// LSTM_11553462026470
// MI455X (gfx1250) — matrix operand order repaired and checked by reading, not compiled, not run on hardware
//
#include <hip/hip_runtime.h>
#include <hip/hip_bf16.h>

// ---------------------------------------------------------------------------
// LSTM forward on gfx1250 (wave32, v_wmma_f32_16x16x32_bf16).
//
// Phase 1 (parallel):  pre[t] = x_t @ W_x + b   (M=32768 GEMM, off the
//                      critical path; only if ws_size fits the 512MB buffer)
// Phase 2 (persistent): single kernel, 64 blocks, all 512 steps with a
//                      device-scope atomic barrier between steps.
//                      - recurrent weights staged ONCE into LDS (<=192KB of
//                        the 320KB/WGP) -> per-step B-tiles come from
//                        ds_load_b128, zero L2 traffic on the serial chain
//                      - cell state c lives in VGPRs the whole sequence
//                      - h double-buffered through L2 in bf16
//                      - 4 gate accumulators per wave => gate fusion in-reg
// Phase 3:             out = h_T @ W_out + b_out
// ---------------------------------------------------------------------------

typedef __attribute__((ext_vector_type(16))) __bf16 v16bf;
typedef __attribute__((ext_vector_type(8)))  __bf16 v8bf;
typedef __attribute__((ext_vector_type(8)))  float  v8f;
typedef __attribute__((ext_vector_type(4)))  float  v4f;

#define LSTM_IN     512
#define LSTM_HID    1024
#define LSTM_OUT    512
#define LSTM_B      64
#define LSTM_T      512
#define LSTM_K      (LSTM_IN + LSTM_HID)   // 1536
#define LSTM_G      (4 * LSTM_HID)         // 4096

// ws layout (bytes, 256B-aligned)
#define WS_WT      0u                       // 4096*1536*2 = 12,582,912
#define WS_WTOUT   12582912u                //  512*1024*2 =  1,048,576
#define WS_H0      13631488u                //   64*1024*2 =    131,072
#define WS_H1      13762560u                //   64*1024*2 =    131,072
#define WS_CTR     13893632u                //  256 (one unsigned + pad)
#define WS_PRE     13893888u                // 512*64*4*4*32*8*4 = 536,870,912
#define WS_NEED_PRE (13893888ull + 536870912ull)

__device__ __forceinline__ float dev_sigmoid(float x) {
    return 1.0f / (1.0f + __expf(-x));
}
__device__ __forceinline__ float dev_tanh(float x) {
    return 1.0f - 2.0f / (__expf(2.0f * x) + 1.0f);   // v_exp_f32 based
}

// ---------------------------------------------------------------------------
// Prep: W -> Wt bf16 [4096][1536] (N-major), W_out -> WtOut bf16 [512][1024],
//       h0 = 0, barrier counter = 0.
// ---------------------------------------------------------------------------
__global__ __launch_bounds__(256) void lstm_prep(
    const float* __restrict__ W, const float* __restrict__ Wout,
    __bf16* __restrict__ Wt, __bf16* __restrict__ WtOut,
    __bf16* __restrict__ h0, unsigned* __restrict__ counter)
{
    const long long N1 = (long long)LSTM_K * LSTM_G;
    const long long N2 = (long long)LSTM_HID * LSTM_OUT;
    const long long N3 = (long long)LSTM_B * LSTM_HID;
    const long long stride = (long long)gridDim.x * blockDim.x;
    const long long tid = (long long)blockIdx.x * blockDim.x + threadIdx.x;

    if (tid == 0) *counter = 0u;
    for (long long i = tid; i < N1; i += stride) {
        long long k = i / LSTM_G, n = i % LSTM_G;
        Wt[n * LSTM_K + k] = (__bf16)W[i];
    }
    for (long long i = tid; i < N2; i += stride) {
        long long k = i / LSTM_OUT, n = i % LSTM_OUT;
        WtOut[n * LSTM_HID + k] = (__bf16)Wout[i];
    }
    for (long long i = tid; i < N3; i += stride) h0[i] = (__bf16)0.0f;
}

// ---------------------------------------------------------------------------
// Phase 1: pre[t][jtile][mtile][g][lane][v] = (x_t @ W_x + b) D-tiles,
// stored in WMMA accumulator layout so phase 2 seeds acc with v8f loads.
// Grid: 512*64 blocks x 128 threads; wave = mtile.
// ---------------------------------------------------------------------------
__global__ __launch_bounds__(128) void lstm_xw(
    const float*  __restrict__ x,     // [64][512][512]
    const __bf16* __restrict__ Wt,    // [4096][1536]
    const float*  __restrict__ bias,  // [4096]
    float*        __restrict__ pre)
{
    const int lane  = threadIdx.x & 31;
    const int half  = lane >> 4;
    const int l15   = lane & 15;
    const int mtile = threadIdx.x >> 5;        // 0..3
    const int t     = blockIdx.x >> 6;         // 0..511
    const int jtile = blockIdx.x & 63;         // 0..63

    const int arow = mtile * 16 + l15;
    const int ncol = jtile * 16 + l15;

    v8f acc[4];
#pragma unroll
    for (int g = 0; g < 4; ++g) {
        const float bv = bias[g * LSTM_HID + ncol];
#pragma unroll
        for (int v = 0; v < 8; ++v) acc[g][v] = bv;
    }

    const float* xrow = x + ((long long)arow * LSTM_T + t) * LSTM_IN;

    for (int kb = 0; kb < LSTM_IN; kb += 32) {
        v16bf a;
        const int k0 = kb + 16 * half;
        const int k1 = kb + 16 * half + 8;
        const v4f f0 = *(const v4f*)(xrow + k0);
        const v4f f1 = *(const v4f*)(xrow + k0 + 4);
        const v4f f2 = *(const v4f*)(xrow + k1);
        const v4f f3 = *(const v4f*)(xrow + k1 + 4);
#pragma unroll
        for (int e = 0; e < 4; ++e) {
            a[e]      = (__bf16)f0[e];
            a[4 + e]  = (__bf16)f1[e];
            a[8 + e]  = (__bf16)f2[e];
            a[12 + e] = (__bf16)f3[e];
        }
        const int kB = kb + 16 * half;
#pragma unroll
        for (int g = 0; g < 4; ++g) {
            const v16bf bv =
                *(const v16bf*)(Wt + (long long)(g * LSTM_HID + ncol) * LSTM_K + kB);
            acc[g] = __builtin_amdgcn_wmma_f32_16x16x32_bf16(
                false, a, false, bv, (short)0, acc[g], false, false);
        }
    }

    const size_t base =
        ((((size_t)t * 64 + jtile) * 4 + mtile) * 4) * 256 + (size_t)lane * 8;
#pragma unroll
    for (int g = 0; g < 4; ++g)
        *(v8f*)(pre + base + (size_t)g * 256) = acc[g];
}

// ---------------------------------------------------------------------------
// Phase 2 body, templated on KSTART (0 = fused fallback, 512 = pre path) so
// the LDS span is a compile-time constant.
// ---------------------------------------------------------------------------
template <int KSTART>
__device__ __forceinline__ void lstm_steps(
    const float*  __restrict__ x,
    const __bf16* __restrict__ Wt,
    const float*  __restrict__ bias,
    const float*  __restrict__ pre,
    __bf16* __restrict__ h0, __bf16* __restrict__ h1,
    unsigned* __restrict__ counter,
    __bf16* ldsB)
{
    constexpr int KSPAN = LSTM_K - KSTART;    // 1024 (pre) or 1536 (fused)

    const int lane  = threadIdx.x & 31;
    const int half  = lane >> 4;
    const int l15   = lane & 15;
    const int mtile = threadIdx.x >> 5;       // 0..3
    const int jtile = blockIdx.x;             // 0..63

    const int arow = mtile * 16 + l15;
    const int ncol = jtile * 16 + l15;

    // ---- stage this block's recurrent weight slice into LDS (once) ----
    // ldsB[r][k] with r = g*16 + c (c = column within jtile), k in [0,KSPAN)
    {
        constexpr int CPR    = KSPAN / 8;     // v8bf chunks per row
        constexpr int CHUNKS = 64 * CPR;
        for (int idx = threadIdx.x; idx < CHUNKS; idx += 128) {
            const int r  = idx / CPR;
            const int ck = idx - r * CPR;
            const int g  = r >> 4, cc = r & 15;
            const long long n = (long long)(g * LSTM_HID + jtile * 16 + cc);
            *(v8bf*)(ldsB + r * KSPAN + ck * 8) =
                *(const v8bf*)(Wt + n * LSTM_K + KSTART + ck * 8);
        }
    }
    __syncthreads();

    v8f creg;                                  // cell state, register-resident
#pragma unroll
    for (int v = 0; v < 8; ++v) creg[v] = 0.0f;

    for (int t = 0; t < LSTM_T; ++t) {
        const __bf16* hin  = (t & 1) ? h1 : h0;
        __bf16*       hout = (t & 1) ? h0 : h1;

        v8f acc[4];
        if (KSTART != 0) {
            const size_t base =
                ((((size_t)t * 64 + jtile) * 4 + mtile) * 4) * 256 +
                (size_t)lane * 8;
#pragma unroll
            for (int g = 0; g < 4; ++g)
                acc[g] = *(const v8f*)(pre + base + (size_t)g * 256);
        } else {
#pragma unroll
            for (int g = 0; g < 4; ++g) {
                const float bv = bias[g * LSTM_HID + ncol];
#pragma unroll
                for (int v = 0; v < 8; ++v) acc[g][v] = bv;
            }
        }

        const float*  xrow = x + ((long long)arow * LSTM_T + t) * LSTM_IN;
        const __bf16* hrow = hin + (long long)arow * LSTM_HID;

        for (int kb = KSTART; kb < LSTM_K; kb += 32) {
            // A tile (16x32 bf16): elems 0..15 at K=kb+16*half+e
            v16bf a;
            const int k0 = kb + 16 * half;
            const int k1 = kb + 16 * half + 8;
            if (kb < LSTM_IN) {                // fused-fallback region only
                const v4f f0 = *(const v4f*)(xrow + k0);
                const v4f f1 = *(const v4f*)(xrow + k0 + 4);
                const v4f f2 = *(const v4f*)(xrow + k1);
                const v4f f3 = *(const v4f*)(xrow + k1 + 4);
#pragma unroll
                for (int e = 0; e < 4; ++e) {
                    a[e]      = (__bf16)f0[e];
                    a[4 + e]  = (__bf16)f1[e];
                    a[8 + e]  = (__bf16)f2[e];
                    a[12 + e] = (__bf16)f3[e];
                }
            } else {
                const v8bf hv0 = *(const v8bf*)(hrow + (k0 - LSTM_IN));
                const v8bf hv1 = *(const v8bf*)(hrow + (k1 - LSTM_IN));
#pragma unroll
                for (int e = 0; e < 8; ++e) { a[e] = hv0[e]; a[8 + e] = hv1[e]; }
            }

            // B tiles from LDS: lane l -> col n=l15, K = kb + 16*half + e
            const int kB = (kb - KSTART) + 16 * half;
#pragma unroll
            for (int g = 0; g < 4; ++g) {
                const v16bf bv =
                    *(const v16bf*)(ldsB + (g * 16 + l15) * KSPAN + kB);
                acc[g] = __builtin_amdgcn_wmma_f32_16x16x32_bf16(
                    false, a, false, bv, (short)0, acc[g], false, false);
            }
        }

        // gate fusion + in-register cell update
#pragma unroll
        for (int v = 0; v < 8; ++v) {
            const int brow = mtile * 16 + half * 8 + v;   // D: m = v + 8*half
            const float fg = dev_sigmoid(acc[0][v]);
            const float ig = dev_sigmoid(acc[1][v]);
            const float gg = dev_tanh   (acc[2][v]);
            const float og = dev_sigmoid(acc[3][v]);
            const float cn = creg[v] * fg + ig * gg;
            creg[v] = cn;
            hout[(long long)brow * LSTM_HID + ncol] = (__bf16)(og * dev_tanh(cn));
        }

        // ---- device-wide barrier between steps (all 64 blocks resident) ----
        __syncthreads();
        __threadfence();                              // release h stores
        if (threadIdx.x == 0) {
            const unsigned tgt = 64u * (unsigned)(t + 1);
            __hip_atomic_fetch_add(counter, 1u, __ATOMIC_RELEASE,
                                   __HIP_MEMORY_SCOPE_AGENT);
            while (__hip_atomic_load(counter, __ATOMIC_ACQUIRE,
                                     __HIP_MEMORY_SCOPE_AGENT) < tgt)
                __builtin_amdgcn_s_sleep(1);
        }
        __syncthreads();                              // fan acquire out
    }
}

__global__ __launch_bounds__(128) void lstm_recurrent(
    const float*  __restrict__ x,
    const __bf16* __restrict__ Wt,
    const float*  __restrict__ bias,
    const float*  __restrict__ pre,   // tile-packed preacts or nullptr
    __bf16*       __restrict__ h0,
    __bf16*       __restrict__ h1,
    unsigned*     __restrict__ counter)
{
    extern __shared__ __bf16 ldsB[];   // 128KB (pre) / 192KB (fused) of Wt
    if (pre)
        lstm_steps<LSTM_IN>(x, Wt, bias, pre, h0, h1, counter, ldsB);
    else
        lstm_steps<0>(x, Wt, bias, pre, h0, h1, counter, ldsB);
}

// ---------------------------------------------------------------------------
// Phase 3: out = h_T @ W_out + b_out.  Grid: 32 blocks x 128 threads.
// ---------------------------------------------------------------------------
__global__ __launch_bounds__(128) void lstm_proj(
    const __bf16* __restrict__ h,      // [64][1024]
    const __bf16* __restrict__ WtOut,  // [512][1024]
    const float*  __restrict__ bout,   // [512]
    float*        __restrict__ out)    // [64][512]
{
    const int lane  = threadIdx.x & 31;
    const int half  = lane >> 4;
    const int l15   = lane & 15;
    const int mtile = threadIdx.x >> 5;
    const int ntile = blockIdx.x;

    const int arow = mtile * 16 + l15;
    const int ncol = ntile * 16 + l15;

    v8f acc;
    const float bv = bout[ncol];
#pragma unroll
    for (int v = 0; v < 8; ++v) acc[v] = bv;

    const __bf16* hrow = h + (long long)arow * LSTM_HID;

    for (int kb = 0; kb < LSTM_HID; kb += 32) {
        v16bf a;
        const int k0 = kb + 16 * half;
        const int k1 = kb + 16 * half + 8;
        const v8bf hv0 = *(const v8bf*)(hrow + k0);
        const v8bf hv1 = *(const v8bf*)(hrow + k1);
#pragma unroll
        for (int e = 0; e < 8; ++e) { a[e] = hv0[e]; a[8 + e] = hv1[e]; }

        const v16bf bvv =
            *(const v16bf*)(WtOut + (long long)ncol * LSTM_HID + kb + 16 * half);
        acc = __builtin_amdgcn_wmma_f32_16x16x32_bf16(
            false, a, false, bvv, (short)0, acc, false, false);
    }

#pragma unroll
    for (int v = 0; v < 8; ++v) {
        const int brow = mtile * 16 + half * 8 + v;
        out[(long long)brow * LSTM_OUT + ncol] = acc[v];
    }
}

// ---------------------------------------------------------------------------
// Host orchestration: 3-4 launches total, all on `stream`, graph-capture safe.
// ---------------------------------------------------------------------------
extern "C" void kernel_launch(void* const* d_in, const int* in_sizes, int n_in,
                              void* d_out, int out_size, void* d_ws, size_t ws_size,
                              hipStream_t stream) {
    const float* x    = (const float*)d_in[0];
    const float* W    = (const float*)d_in[1];
    const float* b    = (const float*)d_in[2];
    const float* Wout = (const float*)d_in[3];
    const float* bout = (const float*)d_in[4];
    float* out = (float*)d_out;

    char* ws = (char*)d_ws;
    __bf16*   Wt    = (__bf16*)(ws + WS_WT);
    __bf16*   WtOut = (__bf16*)(ws + WS_WTOUT);
    __bf16*   h0    = (__bf16*)(ws + WS_H0);
    __bf16*   h1    = (__bf16*)(ws + WS_H1);
    unsigned* ctr   = (unsigned*)(ws + WS_CTR);
    // x@W_x precompute buffer only if workspace is large enough
    // (branch is deterministic: ws_size is fixed for a given harness).
    float* pre = (ws_size >= WS_NEED_PRE) ? (float*)(ws + WS_PRE) : nullptr;

    lstm_prep<<<1024, 256, 0, stream>>>(W, Wout, Wt, WtOut, h0, ctr);

    if (pre) {
        // Parallel phase: all 512 timesteps' input contributions at once.
        lstm_xw<<<LSTM_T * 64, 128, 0, stream>>>(x, Wt, b, pre);
    }

    // Persistent recurrent kernel: all 512 steps, device-wide barrier between,
    // recurrent weights LDS-resident (<=192KB of 320KB/WGP).
    const size_t shmem = (size_t)64 * (pre ? LSTM_HID : LSTM_K) * sizeof(__bf16);
    lstm_recurrent<<<64, 128, shmem, stream>>>(x, Wt, b, pre, h0, h1, ctr);

    // t=511 (odd) wrote h0 -> final hidden state is in h0.
    lstm_proj<<<32, 128, 0, stream>>>(h0, WtOut, bout, out);
}
